// GINTrain_80633716015171
// MI455X (gfx1250) — compile-verified, matrix operand order checked by reading, not run on hardware
//
#include <hip/hip_runtime.h>
#include <hip/hip_bf16.h>

// ---------------- problem constants (from reference) ----------------
#define N_NODES   131072
#define N_EDGES   262144
#define N_GRAPHS  4096
#define EMB       300
#define N_LAYER   5
#define SELF_LOOP 4
#define BN_EPS    1e-5f
// padded GEMM dims (multiples of 32/64 for WMMA + 4-tile waves)
#define KP1  320    // padded K of GEMM1 (300 -> 320)
#define NP1  640    // padded N of GEMM1 / K of GEMM2 (600 -> 640)
#define KP2  640
#define NP2  320    // padded N of GEMM2 (300 -> 320)
#define HEADH 150
// LDS B-tile row stride in elements (80B = 20 banks -> 16 conflict-free rows)
#define BSTRIDE 40
#define BTILE   (64 * BSTRIDE)

typedef __attribute__((ext_vector_type(16))) __bf16 v16bf;
typedef __attribute__((ext_vector_type(8)))  __bf16 v8bf;
typedef __attribute__((ext_vector_type(8)))  float  v8f;

// ---------------- small utility kernels ----------------
__global__ void zero_f32(float* __restrict__ p, long long n) {
  long long i = (long long)blockIdx.x * blockDim.x + threadIdx.x;
  if (i < n) p[i] = 0.f;
}

// h = atom_emb0[x0] + atom_emb1[x1]
__global__ void embed_nodes(const int* __restrict__ x,
                            const float* __restrict__ e0,
                            const float* __restrict__ e1,
                            float* __restrict__ h) {
  long long gid = (long long)blockIdx.x * blockDim.x + threadIdx.x;
  long long total = (long long)N_NODES * EMB;
  if (gid >= total) return;
  int node = (int)(gid / EMB);
  int f    = (int)(gid - (long long)node * EMB);
  int x0 = x[2 * node + 0];
  int x1 = x[2 * node + 1];
  h[gid] = e0[x0 * EMB + f] + e1[x1 * EMB + f];
}

// agg[dst] += h[src] + bond_emb0[t0] + bond_emb1[t1]   (incl. self loops)
__global__ void scatter_msg(const float* __restrict__ h,
                            const int* __restrict__ eidx,
                            const int* __restrict__ eattr,
                            const float* __restrict__ be0,
                            const float* __restrict__ be1,
                            float* __restrict__ agg) {
  long long gid = (long long)blockIdx.x * blockDim.x + threadIdx.x;
  long long total = (long long)(N_EDGES + N_NODES) * EMB;
  if (gid >= total) return;
  int e = (int)(gid / EMB);
  int f = (int)(gid - (long long)e * EMB);
  int src, dst, t0, t1;
  if (e < N_EDGES) {
    src = eidx[e];
    dst = eidx[N_EDGES + e];
    t0  = eattr[2 * e + 0];
    t1  = eattr[2 * e + 1];
  } else {
    src = dst = e - N_EDGES;
    t0 = SELF_LOOP; t1 = 0;
  }
  float v = h[(long long)src * EMB + f] + be0[t0 * EMB + f] + be1[t1 * EMB + f];
  atomicAdd(&agg[(long long)dst * EMB + f], v);
}

// agg (f32 [N x 300]) -> aggBF (bf16 [N x 320], zero padded)
__global__ void cvt_agg_bf16(const float* __restrict__ agg,
                             __bf16* __restrict__ aggBF) {
  long long gid = (long long)blockIdx.x * blockDim.x + threadIdx.x;
  long long total = (long long)N_NODES * KP1;
  if (gid >= total) return;
  int row = (int)(gid / KP1);
  int k   = (int)(gid - (long long)row * KP1);
  aggBF[gid] = (k < EMB) ? (__bf16)agg[(long long)row * EMB + k] : (__bf16)0.f;
}

// transpose + convert weight: out[n*KP + k] = W[k*N + n] (bf16, zero padded)
__global__ void prep_weight(const float* __restrict__ W, __bf16* __restrict__ out,
                            int K, int N, int KP, int NP) {
  long long gid = (long long)blockIdx.x * blockDim.x + threadIdx.x;
  long long total = (long long)NP * KP;
  if (gid >= total) return;
  int n = (int)(gid / KP);
  int k = (int)(gid - (long long)n * KP);
  out[gid] = (k < K && n < N) ? (__bf16)W[(long long)k * N + n] : (__bf16)0.f;
}

// ---------------- WMMA bf16 fragment helpers ----------------
// 16-bit A/B fragment layout (ISA 7.12.2): lanes 0-15 hold K=base..base+7 and
// base+16..base+23; lanes 16-31 hold K=base+8..15 and base+24..31.
__device__ __forceinline__ v16bf pack_frag(v8bf lo, v8bf hi) {
  v16bf r;
#pragma unroll
  for (int j = 0; j < 8; ++j) { r[j] = lo[j]; r[j + 8] = hi[j]; }
  return r;
}

// Core tile loop: block = 256 threads = 8 waves stacked along M (128 rows),
// each wave owns a 16x64 strip (4 accumulators). The 64x32 B tile is staged
// in LDS (double-buffered) with GLOBAL_LOAD_ASYNC_TO_LDS_B128 (ASYNCcnt path,
// no VGPR bounce); per-wave fragment reads come back as ds_load_b128. The
// next tile's async copy is issued before the current tile's WMMAs, so the
// steady-state K loop overlaps DMA, VALU and matrix work.
template <int KP>
__device__ __forceinline__ void wmma_strip(const __bf16* __restrict__ A,
                                           const __bf16* __restrict__ B,
                                           __bf16* __restrict__ Bs,  // [2*BTILE]
                                           int m0, int n0, v8f acc[4]) {
  const int tid  = threadIdx.x;
  const int lane = tid & 31;
  const int half = lane >> 4;
  const int lm   = lane & 15;
  const int srow   = tid >> 2;          // 0..63  (n within tile)
  const int schunk = (tid & 3) * 8;     // 0,8,16,24 (k within tile)
  const __bf16* brow = B + (long long)(n0 + srow) * KP + schunk;
  const __bf16* arow = A + (long long)(m0 + lm) * KP;

  // this thread's 16B staging slot as an LDS-relative byte offset
  unsigned ldsSlot =
      (unsigned)(uintptr_t)(__attribute__((address_space(3))) __bf16*)Bs +
      (unsigned)((srow * BSTRIDE + schunk) * 2);

  // async-stage first B tile into buffer 0 (global -> LDS DMA)
  asm volatile("global_load_async_to_lds_b128 %0, %1, off"
               :: "v"(ldsSlot), "v"(brow) : "memory");
  asm volatile("s_wait_asynccnt 0" ::: "memory");
  __syncthreads();

  int buf = 0;
#pragma unroll 2
  for (int kb = 0; kb + 32 < KP; kb += 32) {
    // kick off DMA of the next B tile into the other buffer (its previous
    // readers were fenced by the barrier at the end of the last iteration)
    asm volatile("global_load_async_to_lds_b128 %0, %1, off"
                 :: "v"(ldsSlot + (unsigned)((buf ^ 1) * BTILE * 2)),
                    "v"(brow + kb + 32) : "memory");
    if (kb + 64 < KP)
      __builtin_prefetch(arow + kb + 64, 0, 3);   // near-cache A prefetch

    // A fragment straight from global (rows are private to this wave)
    v16bf a = pack_frag(*(const v8bf*)(arow + kb + half * 8),
                        *(const v8bf*)(arow + kb + 16 + half * 8));
    // batch all four B-fragment LDS reads, then the four WMMAs
    const __bf16* bbase = Bs + buf * BTILE;
    v16bf bf[4];
#pragma unroll
    for (int t = 0; t < 4; ++t) {
      const __bf16* p = bbase + (t * 16 + lm) * BSTRIDE + half * 8;
      bf[t] = pack_frag(*(const v8bf*)(p), *(const v8bf*)(p + 16));
    }
#pragma unroll
    for (int t = 0; t < 4; ++t)
      acc[t] = __builtin_amdgcn_wmma_f32_16x16x32_bf16(
          false, a, false, bf[t], (short)0, acc[t], false, false);

    asm volatile("s_wait_asynccnt 0" ::: "memory");
    __syncthreads();
    buf ^= 1;
  }
  // peeled last K step (no staging, no barrier)
  {
    const int kb = KP - 32;
    v16bf a = pack_frag(*(const v8bf*)(arow + kb + half * 8),
                        *(const v8bf*)(arow + kb + 16 + half * 8));
    const __bf16* bbase = Bs + buf * BTILE;
    v16bf bf[4];
#pragma unroll
    for (int t = 0; t < 4; ++t) {
      const __bf16* p = bbase + (t * 16 + lm) * BSTRIDE + half * 8;
      bf[t] = pack_frag(*(const v8bf*)(p), *(const v8bf*)(p + 16));
    }
#pragma unroll
    for (int t = 0; t < 4; ++t)
      acc[t] = __builtin_amdgcn_wmma_f32_16x16x32_bf16(
          false, a, false, bf[t], (short)0, acc[t], false, false);
  }
}

// ---------------- GEMM1: hidBF = relu(aggBF @ W1 + b1)  (bf16 out) --------
template <int KP>
__global__ __launch_bounds__(256) void gemm_bf16_relu(
    const __bf16* __restrict__ A,   // [M x KP]
    const __bf16* __restrict__ B,   // [NP x KP] (transposed weight)
    const float* __restrict__ bias, // [Nreal]
    __bf16* __restrict__ out,       // [M x NP]
    int NP, int Nreal) {
  __shared__ __bf16 Bs[2 * BTILE];
  const int lane = threadIdx.x & 31;
  const int half = lane >> 4;
  const int lm   = lane & 15;
  const int m0 = (blockIdx.x * 8 + (threadIdx.x >> 5)) * 16;
  const int n0 = blockIdx.y * 64;
  v8f acc[4] = {};
  wmma_strip<KP>(A, B, Bs, m0, n0, acc);
  // C/D layout: lane holds N = n0+t*16+(lane&15), M = m0+half*8+r
#pragma unroll
  for (int t = 0; t < 4; ++t) {
    int n = n0 + t * 16 + lm;
    float bi = (n < Nreal) ? bias[n] : 0.f;
#pragma unroll
    for (int r = 0; r < 8; ++r) {
      int m = m0 + half * 8 + r;
      float v = acc[t][r] + bi;
      v = v > 0.f ? v : 0.f;            // pad cols stay exactly zero
      out[(long long)m * NP + n] = (__bf16)v;
    }
  }
}

// ---------------- GEMM2: h2 = hidBF @ W2 + b2  (f32 out, no act) ----------
template <int KP>
__global__ __launch_bounds__(256) void gemm_bf16_f32out(
    const __bf16* __restrict__ A,   // [M x KP]
    const __bf16* __restrict__ B,   // [NP x KP]
    const float* __restrict__ bias, // [Nreal]
    float* __restrict__ out,        // [M x Nreal]
    int Nreal) {
  __shared__ __bf16 Bs[2 * BTILE];
  const int lane = threadIdx.x & 31;
  const int half = lane >> 4;
  const int lm   = lane & 15;
  const int m0 = (blockIdx.x * 8 + (threadIdx.x >> 5)) * 16;
  const int n0 = blockIdx.y * 64;
  v8f acc[4] = {};
  wmma_strip<KP>(A, B, Bs, m0, n0, acc);
#pragma unroll
  for (int t = 0; t < 4; ++t) {
    int n = n0 + t * 16 + lm;
    if (n < Nreal) {
      float bi = bias[n];
#pragma unroll
      for (int r = 0; r < 8; ++r) {
        int m = m0 + half * 8 + r;
        out[(long long)m * Nreal + n] = acc[t][r] + bi;
      }
    }
  }
}

// ---------------- BatchNorm (training-mode batch stats) ----------------
__global__ void bn_stats(const float* __restrict__ h2, float* __restrict__ stats) {
  const int ROWS = 256;
  int f = threadIdx.x;
  if (f >= EMB) return;
  int r0 = blockIdx.x * ROWS;
  float s = 0.f, s2 = 0.f;
  for (int r = 0; r < ROWS; ++r) {
    float v = h2[(long long)(r0 + r) * EMB + f];
    s += v; s2 += v * v;
  }
  atomicAdd(&stats[f], s);
  atomicAdd(&stats[EMB + f], s2);
}

__global__ void bn_apply(const float* __restrict__ h2,
                         const float* __restrict__ stats,
                         const float* __restrict__ gamma,
                         const float* __restrict__ beta,
                         float* __restrict__ h, int doRelu) {
  long long gid = (long long)blockIdx.x * blockDim.x + threadIdx.x;
  long long total = (long long)N_NODES * EMB;
  if (gid >= total) return;
  int f = (int)(gid % EMB);
  const float invN = 1.f / (float)N_NODES;
  float mean = stats[f] * invN;
  float var  = stats[EMB + f] * invN - mean * mean;
  float v = (h2[gid] - mean) * rsqrtf(var + BN_EPS) * gamma[f] + beta[f];
  if (doRelu) v = fmaxf(v, 0.f);
  h[gid] = v;
}

// ---------------- head ----------------
__global__ void last_idx_k(const int* __restrict__ batch, int* __restrict__ li) {
  int i = blockIdx.x * blockDim.x + threadIdx.x;
  if (i < N_NODES) atomicMax(&li[batch[i]], i);
}

__global__ void head1(const float* __restrict__ h, const int* __restrict__ li,
                      const float* __restrict__ W, const float* __restrict__ b,
                      float* __restrict__ o) {
  int g = blockIdx.x;
  int j = threadIdx.x;
  if (j >= HEADH) return;
  const float* row = h + (long long)li[g] * EMB;
  float s = b[j];
  for (int k = 0; k < EMB; ++k) s += row[k] * W[k * HEADH + j];
  s = s > 0.f ? s : (expf(s) - 1.f);   // ELU(alpha=1)
  o[(long long)g * HEADH + j] = s;
}

__global__ void head2(const float* __restrict__ t, const float* __restrict__ W2,
                      const float* __restrict__ b2, float* __restrict__ out) {
  int g = blockIdx.x * blockDim.x + threadIdx.x;
  if (g >= N_GRAPHS) return;
  float s = b2[0];
  for (int k = 0; k < HEADH; ++k) s += t[(long long)g * HEADH + k] * W2[k];
  out[g] = s;
}

// ---------------- launcher ----------------
static inline int gdiv(long long a, int b) { return (int)((a + b - 1) / b); }

extern "C" void kernel_launch(void* const* d_in, const int* in_sizes, int n_in,
                              void* d_out, int out_size, void* d_ws, size_t ws_size,
                              hipStream_t stream) {
  const int*   x     = (const int*)d_in[0];
  const int*   eidx  = (const int*)d_in[1];
  const int*   eattr = (const int*)d_in[2];
  const int*   batch = (const int*)d_in[3];
  // d_in[4] = num_graphs (scalar; compile-time N_GRAPHS used)
  const float* aemb0 = (const float*)d_in[5];
  const float* aemb1 = (const float*)d_in[6];
  const float* bemb0 = (const float*)d_in[7];   // [L,6,300]
  const float* bemb1 = (const float*)d_in[8];   // [L,3,300]
  const float* W1    = (const float*)d_in[9];   // [L,300,600]
  const float* b1    = (const float*)d_in[10];  // [L,600]
  const float* W2    = (const float*)d_in[11];  // [L,600,300]
  const float* b2    = (const float*)d_in[12];  // [L,300]
  const float* gam   = (const float*)d_in[13];
  const float* bet   = (const float*)d_in[14];
  const float* hW1   = (const float*)d_in[15];
  const float* hb1   = (const float*)d_in[16];
  const float* hW2   = (const float*)d_in[17];
  const float* hb2   = (const float*)d_in[18];
  float* out = (float*)d_out;

  // carve workspace
  char* ws = (char*)d_ws;
  size_t off = 0;
  auto carve = [&](size_t bytes) -> char* {
    char* p = ws + off;
    off += (bytes + 255) & ~(size_t)255;
    return p;
  };
  float*  h     = (float*) carve(sizeof(float)  * (size_t)N_NODES * EMB);
  float*  agg   = (float*) carve(sizeof(float)  * (size_t)N_NODES * EMB); // aliased as h2
  __bf16* aggBF = (__bf16*)carve(sizeof(__bf16) * (size_t)N_NODES * KP1);
  __bf16* hidBF = (__bf16*)carve(sizeof(__bf16) * (size_t)N_NODES * NP1);
  __bf16* W1T   = (__bf16*)carve(sizeof(__bf16) * (size_t)NP1 * KP1);
  __bf16* W2T   = (__bf16*)carve(sizeof(__bf16) * (size_t)NP2 * KP2);
  float*  stats = (float*) carve(sizeof(float) * 640);
  int*    lidx  = (int*)   carve(sizeof(int) * N_GRAPHS);
  float*  hhid  = (float*) carve(sizeof(float) * (size_t)N_GRAPHS * HEADH);
  float* h2 = agg;  // agg is dead after cvt_agg_bf16; reuse for GEMM2 output

  const int T = 256;
  const long long nEl   = (long long)N_NODES * EMB;
  const long long msgEl = (long long)(N_EDGES + N_NODES) * EMB;

  // node embedding
  embed_nodes<<<gdiv(nEl, T), T, 0, stream>>>(x, aemb0, aemb1, h);

  for (int l = 0; l < N_LAYER; ++l) {
    // message passing: agg = segment_sum(h[src] + edge_emb, dst)
    zero_f32<<<gdiv(nEl, T), T, 0, stream>>>(agg, nEl);
    scatter_msg<<<gdiv(msgEl, T), T, 0, stream>>>(
        h, eidx, eattr, bemb0 + (size_t)l * 6 * EMB, bemb1 + (size_t)l * 3 * EMB, agg);

    // convert activations and weights to padded bf16
    cvt_agg_bf16<<<gdiv((long long)N_NODES * KP1, T), T, 0, stream>>>(agg, aggBF);
    prep_weight<<<gdiv((long long)NP1 * KP1, T), T, 0, stream>>>(
        W1 + (size_t)l * EMB * 2 * EMB, W1T, EMB, 2 * EMB, KP1, NP1);
    prep_weight<<<gdiv((long long)NP2 * KP2, T), T, 0, stream>>>(
        W2 + (size_t)l * 2 * EMB * EMB, W2T, 2 * EMB, EMB, KP2, NP2);

    // GEMM1 (WMMA bf16 + async-LDS B staging): hidBF = relu(agg @ W1 + b1)
    dim3 g1(N_NODES / 128, NP1 / 64);
    gemm_bf16_relu<KP1><<<g1, 256, 0, stream>>>(aggBF, W1T,
                                                b1 + (size_t)l * 2 * EMB,
                                                hidBF, NP1, 2 * EMB);

    // GEMM2 (WMMA bf16 + async-LDS B staging): h2 = hid @ W2 + b2 (f32)
    dim3 g2(N_NODES / 128, NP2 / 64);
    gemm_bf16_f32out<KP2><<<g2, 256, 0, stream>>>(hidBF, W2T,
                                                  b2 + (size_t)l * EMB,
                                                  h2, EMB);

    // batchnorm (+ relu except last layer)
    zero_f32<<<gdiv((long long)640, T), T, 0, stream>>>(stats, 640);
    bn_stats<<<N_NODES / 256, 320, 0, stream>>>(h2, stats);
    bn_apply<<<gdiv(nEl, T), T, 0, stream>>>(h2, stats, gam + (size_t)l * EMB,
                                             bet + (size_t)l * EMB, h,
                                             (l < N_LAYER - 1) ? 1 : 0);
  }

  // head: super node = last node per graph (batch sorted, indices >= 0)
  zero_f32<<<gdiv((long long)N_GRAPHS, T), T, 0, stream>>>((float*)lidx, N_GRAPHS);
  last_idx_k<<<gdiv((long long)N_NODES, T), T, 0, stream>>>(batch, lidx);
  head1<<<N_GRAPHS, 160, 0, stream>>>(h, lidx, hW1, hb1, hhid);
  head2<<<gdiv((long long)N_GRAPHS, T), T, 0, stream>>>(hhid, hW2, hb2, out);
}
